// ResNetBlock_79353815761241
// MI455X (gfx1250) — matrix operand order repaired and checked by reading, not compiled, not run on hardware
//
#include <hip/hip_runtime.h>
#include <stdint.h>

// ---------------------------------------------------------------------------
// ResNet intrinsic-mesh-conv block for MI455X (gfx1250, wave32, WMMA).
// conv_dirac is recast as GEMM [N x 1280] @ Wbig[1280 x 256] with
// Wbig[(r,a,d),(k,t)] = W[t, r, (a-k) mod 8, d]   (rotation folded into B).
// ---------------------------------------------------------------------------

typedef __attribute__((ext_vector_type(16))) _Float16 v16h;
typedef __attribute__((ext_vector_type(8)))  float    v8f;

#define RR 5
#define AA 8
#define DD 32
#define TT 32
#define KKROT 8                  // rotations (A / ROT_DELTA)
#define KDIM (RR*AA*DD)          // 1280  (GEMM K)
#define CDIM (KKROT*TT)          // 256   (GEMM N)
#define MTILE 32                 // vertices per block
#define BN_EPS 1e-3f

// LDS carve (bytes)
#define A_BYTES    (MTILE*KDIM*2)        // 81920 : f16 interp panel
#define OUT_BYTES  (MTILE*KKROT*TT*4)    // 32768 : f32 conv outputs [v][k][t]
#define NORM_BYTES (MTILE*KKROT*4)       // 1024
#define BK_BYTES   (MTILE*4)             // 128
#define LS_BYTES   (64*4)                // 256  : block-partial BN sums
#define SMEM_BYTES (A_BYTES+OUT_BYTES+NORM_BYTES+BK_BYTES+LS_BYTES)

// ---------------------------------------------------------------------------
// Build rotated weight matrix Wbig[p][c], p=(r*8+a)*32+d, c=k*32+t, f16.
// ---------------------------------------------------------------------------
__global__ void prep_weights(const float* __restrict__ W, _Float16* __restrict__ Wbig) {
    int idx = blockIdx.x * blockDim.x + threadIdx.x;
    if (idx >= KDIM * CDIM) return;
    int p  = idx >> 8;          // 0..1279
    int c  = idx & 255;         // 0..255
    int k  = c >> 5, t = c & 31;
    int d  = p & 31;
    int ra = p >> 5;            // 0..39
    int r  = ra >> 3, a = ra & 7;
    int a2 = (a - k) & 7;       // roll(W, k, axis=2)[...,a,...] = W[...,(a-k)%8,...]
    float w = W[(((t * RR + r) * AA + a2) * DD) + d];
    Wbig[(size_t)p * CDIM + c] = (_Float16)w;
}

__global__ void zero_f32(float* __restrict__ p, int n) {
    int i = blockIdx.x * blockDim.x + threadIdx.x;
    if (i < n) p[i] = 0.f;
}

// ---------------------------------------------------------------------------
// Fused: gather + barycentric interp -> LDS f16 panel -> WMMA GEMM ->
// bias -> angular max-pool -> write [N,32] + BN partial sums (atomics).
// One block = 32 vertices, 256 threads = 8 waves; wave w owns rotation k=w.
// ---------------------------------------------------------------------------
__global__ void conv_dirac_wmma(const float*    __restrict__ src,     // [N,32]
                                const int*      __restrict__ bc_idx,  // [N,5,8,3]
                                const float*    __restrict__ bc_w,    // [N,5,8,3]
                                const _Float16* __restrict__ Wbig,    // [1280,256]
                                const float*    __restrict__ bias,    // [32]
                                float*          __restrict__ outNT,   // [N,32]
                                float*          __restrict__ stats,   // [0:32)=sum [32:64)=sumsq
                                int N)
{
    extern __shared__ char smem[];
    _Float16* A_s  = (_Float16*)smem;
    float*    out_s = (float*)(smem + A_BYTES);
    float*    norm2 = (float*)(smem + A_BYTES + OUT_BYTES);
    int*      bestk = (int*)  (smem + A_BYTES + OUT_BYTES + NORM_BYTES);
    float*    lsum  = (float*)(smem + A_BYTES + OUT_BYTES + NORM_BYTES + BK_BYTES);

    const int tid = threadIdx.x;
    const int v0  = blockIdx.x * MTILE;

    if (tid < 64) lsum[tid] = 0.f;

    // ---- Phase 1: gather + interp, 8 threads per (v,r,a) triple, 4 floats each
    for (int wi = tid; wi < MTILE * RR * AA * 8; wi += blockDim.x) {
        int triple = wi >> 3;
        int sub    = wi & 7;
        int v      = triple / (RR * AA);
        int ra     = triple - v * (RR * AA);
        int dd     = sub * 4;
        union { _Float16 h[4]; uint64_t u; } pk;
        pk.u = 0ull;
        int vg = v0 + v;
        if (vg < N) {
            long g  = ((long)vg * (RR * AA) + ra) * 3;
            int  i0 = bc_idx[g], i1 = bc_idx[g + 1], i2 = bc_idx[g + 2];
            float w0 = bc_w[g], w1 = bc_w[g + 1], w2 = bc_w[g + 2];
            float4 x0 = *(const float4*)(src + (long)i0 * DD + dd);
            float4 x1 = *(const float4*)(src + (long)i1 * DD + dd);
            float4 x2 = *(const float4*)(src + (long)i2 * DD + dd);
            pk.h[0] = (_Float16)(w0 * x0.x + w1 * x1.x + w2 * x2.x);
            pk.h[1] = (_Float16)(w0 * x0.y + w1 * x1.y + w2 * x2.y);
            pk.h[2] = (_Float16)(w0 * x0.z + w1 * x1.z + w2 * x2.z);
            pk.h[3] = (_Float16)(w0 * x0.w + w1 * x1.w + w2 * x2.w);
        }
        *(uint64_t*)(A_s + v * KDIM + ra * DD + dd) = pk.u;
    }
    __syncthreads();

    // ---- Phase 2: WMMA GEMM. wave wv -> rotation k=wv -> columns [wv*32, wv*32+32)
    const int wv   = tid >> 5;
    const int lane = tid & 31;
    const int nlo  = lane & 15;
    const int sel  = (lane < 16) ? 0 : 8;       // A f16 layout: lanes>=16 hold K {8..15,24..31}
    const int mhi  = (lane >> 4) << 3;          // C layout: lanes>=16 hold rows m+8
    const int n0   = wv * TT;

    v8f c00 = {}, c01 = {}, c10 = {}, c11 = {};
    for (int kk = 0; kk < KDIM / 32; ++kk) {
        const int p0 = kk * 32;
        v16h a0, a1;
        {   // A frag, M-tile 0 (rows 0..15)
            uint4* ap = reinterpret_cast<uint4*>(&a0);
            const _Float16* ab = A_s + nlo * KDIM + p0;
            ap[0] = *(const uint4*)(ab + sel);
            ap[1] = *(const uint4*)(ab + 16 + sel);
        }
        {   // A frag, M-tile 1 (rows 16..31)
            uint4* ap = reinterpret_cast<uint4*>(&a1);
            const _Float16* ab = A_s + (16 + nlo) * KDIM + p0;
            ap[0] = *(const uint4*)(ab + sel);
            ap[1] = *(const uint4*)(ab + 16 + sel);
        }
        // B frags: lane = N column (n0+nlo), 16 halves = K rows in A's order
        const _Float16* bb = Wbig + (size_t)(p0 + sel) * CDIM + n0 + nlo;
        v16h bf0, bf1;
#pragma unroll
        for (int s = 0; s < 16; ++s) {
            const int kr = (s < 8) ? s : s + 8;
            bf0[s] = bb[(size_t)kr * CDIM];
            bf1[s] = bb[(size_t)kr * CDIM + 16];
        }

        c00 = __builtin_amdgcn_wmma_f32_16x16x32_f16(false, a0, false, bf0, (short)0, c00, false, false);
        c01 = __builtin_amdgcn_wmma_f32_16x16x32_f16(false, a0, false, bf1, (short)0, c01, false, false);
        c10 = __builtin_amdgcn_wmma_f32_16x16x32_f16(false, a1, false, bf0, (short)0, c10, false, false);
        c11 = __builtin_amdgcn_wmma_f32_16x16x32_f16(false, a1, false, bf1, (short)0, c11, false, false);
    }

    // ---- spill C + bias into LDS as out_s[v][k][t]
    const float bt0 = bias[nlo];
    const float bt1 = bias[16 + nlo];
#pragma unroll
    for (int j = 0; j < 8; ++j) {
        int m = j + mhi;
        out_s[((m)      * KKROT + wv) * TT + nlo]      = c00[j] + bt0;
        out_s[((m)      * KKROT + wv) * TT + 16 + nlo] = c01[j] + bt1;
        out_s[((16 + m) * KKROT + wv) * TT + nlo]      = c10[j] + bt0;
        out_s[((16 + m) * KKROT + wv) * TT + 16 + nlo] = c11[j] + bt1;
    }
    __syncthreads();

    // ---- Phase 3: angular max-pool. One thread per (v,k) pair (32*8 == 256).
    {
        int v = tid >> 3, k = tid & 7;
        const float* row = out_s + (v * KKROT + k) * TT;
        float s = 0.f;
#pragma unroll
        for (int t = 0; t < TT; ++t) { float x = row[t]; s += x * x; }
        norm2[v * KKROT + k] = s;
    }
    __syncthreads();
    if (tid < MTILE) {
        int best = 0; float bv = norm2[tid * KKROT];
#pragma unroll
        for (int k = 1; k < KKROT; ++k) {
            float x = norm2[tid * KKROT + k];
            if (x > bv) { bv = x; best = k; }
        }
        bestk[tid] = best;
    }
    __syncthreads();

    for (int e = tid; e < MTILE * TT; e += blockDim.x) {
        int v = e >> 5, t = e & 31;
        if (v0 + v < N) {
            float val = out_s[(v * KKROT + bestk[v]) * TT + t];
            outNT[(size_t)(v0 + v) * TT + t] = val;
            atomicAdd(&lsum[t], val);
            atomicAdd(&lsum[32 + t], val * val);
        }
    }
    __syncthreads();
    if (tid < 64) atomicAdd(&stats[tid], lsum[tid]);
}

// ---------------------------------------------------------------------------
// BN finalize: stats[64+t]=scale, stats[96+t]=shift
// ---------------------------------------------------------------------------
__global__ void bn_finalize(float* __restrict__ stats,
                            const float* __restrict__ gamma,
                            const float* __restrict__ beta,
                            float invN)
{
    int t = threadIdx.x;
    if (t < TT) {
        float mean = stats[t] * invN;
        float var  = stats[32 + t] * invN - mean * mean;
        float sc   = gamma[t] * rsqrtf(var + BN_EPS);
        stats[64 + t] = sc;
        stats[96 + t] = beta[t] - mean * sc;
    }
}

__global__ void bn_apply_relu(const float* __restrict__ x,
                              const float* __restrict__ stats,
                              float* __restrict__ y, int n)
{
    int i = blockIdx.x * blockDim.x + threadIdx.x;
    if (i < n) {
        int t = i & 31;
        float v = stats[64 + t] * x[i] + stats[96 + t];
        y[i] = v > 0.f ? v : 0.f;
    }
}

__global__ void bn_residual_relu(const float* __restrict__ x,
                                 const float* __restrict__ stats,
                                 const float* __restrict__ res,
                                 float* __restrict__ y, int n)
{
    int i = blockIdx.x * blockDim.x + threadIdx.x;
    if (i < n) {
        int t = i & 31;
        float v = stats[64 + t] * x[i] + stats[96 + t] + res[i];
        y[i] = v > 0.f ? v : 0.f;
    }
}

// ---------------------------------------------------------------------------
extern "C" void kernel_launch(void* const* d_in, const int* in_sizes, int n_in,
                              void* d_out, int out_size, void* d_ws, size_t ws_size,
                              hipStream_t stream) {
    const float* signal = (const float*)d_in[0];
    const int*   bc_idx = (const int*)  d_in[1];
    const float* bc_w   = (const float*)d_in[2];
    const float* W1     = (const float*)d_in[3];
    const float* b1     = (const float*)d_in[4];
    const float* gamma1 = (const float*)d_in[5];
    const float* beta1  = (const float*)d_in[6];
    const float* W2     = (const float*)d_in[7];
    const float* b2     = (const float*)d_in[8];
    const float* gamma2 = (const float*)d_in[9];
    const float* beta2  = (const float*)d_in[10];
    float* out = (float*)d_out;

    const int N = in_sizes[0] / DD;           // 80000
    const int nElems = N * TT;

    // ---- workspace carve (all offsets 256B aligned)
    char* w = (char*)d_ws;
    const size_t WBIG_BYTES = (size_t)KDIM * CDIM * 2;   // 655360
    _Float16* Wbig1 = (_Float16*)w;            w += WBIG_BYTES;
    _Float16* Wbig2 = (_Float16*)w;            w += WBIG_BYTES;
    float* stats1   = (float*)w;               w += 512;
    float* stats2   = (float*)w;               w += 512;
    float* s1       = (float*)w;               w += (size_t)nElems * 4;
    float* s1n      = (float*)w;               w += (size_t)nElems * 4;
    float* s2       = (float*)w;               w += (size_t)nElems * 4;
    (void)ws_size; (void)n_in; (void)out_size;

    const int convBlocks = (N + MTILE - 1) / MTILE;      // 2500
    const int ewBlocks   = (nElems + 255) / 256;
    const float invN = 1.0f / (float)N;

    // stats must be zeroed every call (deterministic graph replay)
    zero_f32<<<1, 256, 0, stream>>>(stats1, 128);
    zero_f32<<<1, 256, 0, stream>>>(stats2, 128);

    prep_weights<<<(KDIM * CDIM) / 256, 256, 0, stream>>>(W1, Wbig1);
    prep_weights<<<(KDIM * CDIM) / 256, 256, 0, stream>>>(W2, Wbig2);

    // Branch 1: conv -> pool (+ BN partial sums)
    conv_dirac_wmma<<<convBlocks, 256, SMEM_BYTES, stream>>>(
        signal, bc_idx, bc_w, Wbig1, b1, s1, stats1, N);
    bn_finalize<<<1, 32, 0, stream>>>(stats1, gamma1, beta1, invN);
    bn_apply_relu<<<ewBlocks, 256, 0, stream>>>(s1, stats1, s1n, nElems);

    // Branch 2: conv -> pool (+ BN partial sums)
    conv_dirac_wmma<<<convBlocks, 256, SMEM_BYTES, stream>>>(
        s1n, bc_idx, bc_w, Wbig2, b2, s2, stats2, N);
    bn_finalize<<<1, 32, 0, stream>>>(stats2, gamma2, beta2, invN);

    // BN apply + residual + relu
    bn_residual_relu<<<ewBlocks, 256, 0, stream>>>(s2, stats2, signal, out, nElems);
}
